// EnsembleLayer_80341658239521
// MI455X (gfx1250) — matrix operand order repaired and checked by reading, not compiled, not run on hardware
//
#include <hip/hip_runtime.h>
#include <hip/hip_bf16.h>
#include <cstdint>
#include <cstddef>

// Problem constants (match reference)
#define B_N   4096
#define IN_F  1024
#define OUT_F 1024
#define E_N   16

// Tiling
#define KC      128              // K chunk staged in LDS
#define M_TILE  32               // rows per block (2 WMMA row tiles)
#define N_WAVE  32               // cols per wave (2 WMMA col tiles)
#define N_BLOCK 256              // cols per block (8 waves)
#define MAX_TILES (B_N / M_TILE + E_N)   // 144 (covers worst-case padding)

// Workspace layout
//   int region (offsets in ints):
#define OFF_TOTAL  0             // 1 int: number of tiles
#define OFF_COUNTS 16            // E ints: per-expert sample count
#define OFF_TILE_E 64            // MAX_TILES ints: expert of tile
#define OFF_TILE_B 256           // MAX_TILES ints: row base (within expert list)
#define OFF_LIST   512           // E*B ints: per-expert sample index lists
#define WS_INT_COUNT (OFF_LIST + E_N * B_N)            // 66048 ints
#define WS_SPLIT_OFF ((size_t)WS_INT_COUNT * 4)        // 264192 B (512B aligned)
#define W_ELEMS ((size_t)E_N * OUT_F * IN_F)           // 16M elements per matrix
// split region: Wh | Wl | Wph | Wpl, each W_ELEMS bf16 (2B)
#define WS_NEED (WS_SPLIT_OFF + 4 * W_ELEMS * 2)

typedef __attribute__((ext_vector_type(16))) __bf16          v16bf;
typedef __attribute__((ext_vector_type(8)))  float           v8f;

struct U4x2 { uint4 a, b; };     // 32 bytes, bit-castable to v16bf
struct U8w  { unsigned w[8]; };  // 32 bytes, bit-castable to v16bf

// Split 4 fp32 -> packed bf16 hi pair-words + lo pair-words.
// hi = truncate(x) to bf16; lo = truncate(x - hi) to bf16 (captures mantissa
// bits 8..15 -> ~16 effective mantissa bits after 3-term WMMA recombination).
__device__ __forceinline__ void split4_pack(float4 f, uint2& hv, uint2& lv) {
    unsigned u0 = __builtin_bit_cast(unsigned, f.x);
    unsigned u1 = __builtin_bit_cast(unsigned, f.y);
    unsigned u2 = __builtin_bit_cast(unsigned, f.z);
    unsigned u3 = __builtin_bit_cast(unsigned, f.w);
    float r0 = f.x - __builtin_bit_cast(float, u0 & 0xFFFF0000u);
    float r1 = f.y - __builtin_bit_cast(float, u1 & 0xFFFF0000u);
    float r2 = f.z - __builtin_bit_cast(float, u2 & 0xFFFF0000u);
    float r3 = f.w - __builtin_bit_cast(float, u3 & 0xFFFF0000u);
    hv.x = (u0 >> 16) | (u1 & 0xFFFF0000u);
    hv.y = (u2 >> 16) | (u3 & 0xFFFF0000u);
    lv.x = (__builtin_bit_cast(unsigned, r0) >> 16) |
           (__builtin_bit_cast(unsigned, r1) & 0xFFFF0000u);
    lv.y = (__builtin_bit_cast(unsigned, r2) >> 16) |
           (__builtin_bit_cast(unsigned, r3) & 0xFFFF0000u);
}

// In-register split of 16 contiguous fp32 (fallback path, no presplit storage)
__device__ __forceinline__ void split16f(const float* __restrict__ src,
                                         v16bf& bh, v16bf& bl) {
    U8w H, L;
    #pragma unroll
    for (int i = 0; i < 4; ++i) {
        float4 f = *(const float4*)(src + 4 * i);
        uint2 hv, lv;
        split4_pack(f, hv, lv);
        H.w[2 * i + 0] = hv.x; H.w[2 * i + 1] = hv.y;
        L.w[2 * i + 0] = lv.x; L.w[2 * i + 1] = lv.y;
    }
    bh = __builtin_bit_cast(v16bf, H);
    bl = __builtin_bit_cast(v16bf, L);
}

// Load 16 contiguous bf16 halves (32B) as a v16bf
__device__ __forceinline__ v16bf load16bf(const unsigned short* __restrict__ p) {
    U4x2 s;
    s.a = *(const uint4*)p;
    s.b = *(const uint4*)(p + 8);
    return __builtin_bit_cast(v16bf, s);
}

__global__ void ens_zero(int* __restrict__ ws) {
    int t = threadIdx.x;
    if (t < 64) ws[t] = 0;       // clears OFF_TOTAL + OFF_COUNTS region
}

__global__ void ens_route(const int* __restrict__ eidx, int* __restrict__ ws) {
    int i = blockIdx.x * blockDim.x + threadIdx.x;
    if (i < B_N) {
        int e   = eidx[i];
        int pos = atomicAdd(ws + OFF_COUNTS + e, 1);
        ws[OFF_LIST + e * B_N + pos] = i;
    }
}

__global__ void ens_plan(int* __restrict__ ws) {
    if (threadIdx.x == 0 && blockIdx.x == 0) {
        int t = 0;
        for (int e = 0; e < E_N; ++e) {
            int cnt = ws[OFF_COUNTS + e];
            for (int i = 0; i < cnt; i += M_TILE) {
                ws[OFF_TILE_E + t] = e;
                ws[OFF_TILE_B + t] = i;
                ++t;
            }
        }
        ws[OFF_TOTAL] = t;
    }
}

// One-pass W -> (bf16 hi, bf16 lo) planes. Bandwidth bound (~256MB total).
__global__ __launch_bounds__(256)
void ens_split_w(const float4* __restrict__ src, uint2* __restrict__ Hd,
                 uint2* __restrict__ Ld, int n4) {
    int i = blockIdx.x * 256 + threadIdx.x;
    if (i < n4) {
        uint2 hv, lv;
        split4_pack(src[i], hv, lv);
        Hd[i] = hv;
        Ld[i] = lv;
    }
}

template <bool PRESPLIT>
__global__ __launch_bounds__(256)
void ens_gemm(const float* __restrict__ X,  const float* __restrict__ Xp,
              const float* __restrict__ W,  const float* __restrict__ bia,
              const float* __restrict__ Wp, const float* __restrict__ biap,
              const unsigned short* __restrict__ Wh,  const unsigned short* __restrict__ Wl,
              const unsigned short* __restrict__ Wph, const unsigned short* __restrict__ Wpl,
              float* __restrict__ out, const int* __restrict__ ws) {
    const int nTiles = ws[OFF_TOTAL];
    const int ty = blockIdx.y;
    if (ty >= nTiles) return;

    const int e     = ws[OFF_TILE_E + ty];
    const int tbase = ws[OFF_TILE_B + ty];
    const int cnt   = ws[OFF_COUNTS + e];

    const int tid  = threadIdx.x;
    const int lane = tid & 31;
    const int wv   = tid >> 5;                 // wave id 0..7 -> 32-col slice

    __shared__ __attribute__((aligned(16))) unsigned short sAh[2][M_TILE][KC];
    __shared__ __attribute__((aligned(16))) unsigned short sAl[2][M_TILE][KC];
    __shared__ int sRows[M_TILE];

    if (tid < M_TILE) {
        int g = tbase + tid;
        sRows[tid] = (g < cnt) ? ws[OFF_LIST + e * B_N + g] : -1;
    }

    const int col0 = blockIdx.x * N_BLOCK + wv * N_WAVE + (lane & 15);
    const int col1 = col0 + 16;
    const int kb   = (lane < 16) ? 0 : 16;     // B-matrix K half per lane group
    const int akb  = (lane < 16) ? 0 : 16;     // A-matrix K offset per lane group

    // Fallback path row pointers (fp32 W)
    const float* wRow[2][2];
    wRow[0][0] = W  + (size_t)(e * OUT_F + col0) * IN_F;
    wRow[0][1] = W  + (size_t)(e * OUT_F + col1) * IN_F;
    wRow[1][0] = Wp + (size_t)(e * OUT_F + col0) * IN_F;
    wRow[1][1] = Wp + (size_t)(e * OUT_F + col1) * IN_F;

    // Presplit path row pointers (bf16 hi/lo planes)
    const unsigned short* hRow[2][2];
    const unsigned short* lRow[2][2];
    hRow[0][0] = Wh  + (size_t)(e * OUT_F + col0) * IN_F;
    hRow[0][1] = Wh  + (size_t)(e * OUT_F + col1) * IN_F;
    hRow[1][0] = Wph + (size_t)(e * OUT_F + col0) * IN_F;
    hRow[1][1] = Wph + (size_t)(e * OUT_F + col1) * IN_F;
    lRow[0][0] = Wl  + (size_t)(e * OUT_F + col0) * IN_F;
    lRow[0][1] = Wl  + (size_t)(e * OUT_F + col1) * IN_F;
    lRow[1][0] = Wpl + (size_t)(e * OUT_F + col0) * IN_F;
    lRow[1][1] = Wpl + (size_t)(e * OUT_F + col1) * IN_F;

    v8f acc00 = {};   // rows 0..15,  col0 tile
    v8f acc01 = {};   // rows 0..15,  col1 tile
    v8f acc10 = {};   // rows 16..31, col0 tile
    v8f acc11 = {};   // rows 16..31, col1 tile

    for (int k0 = 0; k0 < IN_F; k0 += KC) {
        __syncthreads();
        // Stage A (logits + prior) rows -> LDS as split bf16 hi/lo
        for (int idx = tid; idx < 2 * M_TILE * (KC / 4); idx += 256) {
            int mat = idx / (M_TILE * (KC / 4));
            int rem = idx - mat * (M_TILE * (KC / 4));
            int r   = rem / (KC / 4);
            int kq  = rem - r * (KC / 4);
            int row = sRows[r];
            float4 f = make_float4(0.f, 0.f, 0.f, 0.f);
            if (row >= 0) {
                const float* src = (mat ? Xp : X) + (size_t)row * IN_F + k0 + kq * 4;
                f = *(const float4*)src;
            }
            uint2 hv, lv;
            split4_pack(f, hv, lv);
            *(uint2*)&sAh[mat][r][kq * 4] = hv;
            *(uint2*)&sAl[mat][r][kq * 4] = lv;
        }
        __syncthreads();

        #pragma unroll
        for (int kk = 0; kk < KC; kk += 32) {
            const int koff = k0 + kk + kb;
            #pragma unroll
            for (int mat = 0; mat < 2; ++mat) {
                v16bf bh0, bl0, bh1, bl1;
                if (PRESPLIT) {
                    bh0 = load16bf(hRow[mat][0] + koff);
                    bl0 = load16bf(lRow[mat][0] + koff);
                    bh1 = load16bf(hRow[mat][1] + koff);
                    bl1 = load16bf(lRow[mat][1] + koff);
                } else {
                    split16f(wRow[mat][0] + koff, bh0, bl0);
                    split16f(wRow[mat][1] + koff, bh1, bl1);
                }
                #pragma unroll
                for (int mt = 0; mt < 2; ++mt) {
                    const int r = mt * 16 + (lane & 15);
                    U4x2 sh, sl;
                    sh.a = *(const uint4*)&sAh[mat][r][kk + akb];
                    sh.b = *(const uint4*)&sAh[mat][r][kk + akb + 8];
                    sl.a = *(const uint4*)&sAl[mat][r][kk + akb];
                    sl.b = *(const uint4*)&sAl[mat][r][kk + akb + 8];
                    v16bf ah = __builtin_bit_cast(v16bf, sh);
                    v16bf al = __builtin_bit_cast(v16bf, sl);
                    v8f& a0 = mt ? acc10 : acc00;
                    v8f& a1 = mt ? acc11 : acc01;
                    a0 = __builtin_amdgcn_wmma_f32_16x16x32_bf16(false, ah, false, bh0, (short)0, a0, false, false);
                    a0 = __builtin_amdgcn_wmma_f32_16x16x32_bf16(false, ah, false, bl0, (short)0, a0, false, false);
                    a0 = __builtin_amdgcn_wmma_f32_16x16x32_bf16(false, al, false, bh0, (short)0, a0, false, false);
                    a1 = __builtin_amdgcn_wmma_f32_16x16x32_bf16(false, ah, false, bh1, (short)0, a1, false, false);
                    a1 = __builtin_amdgcn_wmma_f32_16x16x32_bf16(false, ah, false, bl1, (short)0, a1, false, false);
                    a1 = __builtin_amdgcn_wmma_f32_16x16x32_bf16(false, al, false, bh1, (short)0, a1, false, false);
                }
            }
        }
    }

    // Epilogue: bias add + scatter valid rows (POSTERIOR_SCALE = PRIOR_SCALE = 1)
    const float bias0 = bia[e * OUT_F + col0] + biap[e * OUT_F + col0];
    const float bias1 = bia[e * OUT_F + col1] + biap[e * OUT_F + col1];
    const int mrow = (lane >= 16) ? 8 : 0;
    #pragma unroll
    for (int v = 0; v < 8; ++v) {
        int r0 = v + mrow;
        int row0 = sRows[r0];
        if (row0 >= 0) {
            out[(size_t)row0 * OUT_F + col0] = acc00[v] + bias0;
            out[(size_t)row0 * OUT_F + col1] = acc01[v] + bias1;
        }
        int r1 = 16 + v + mrow;
        int row1 = sRows[r1];
        if (row1 >= 0) {
            out[(size_t)row1 * OUT_F + col0] = acc10[v] + bias0;
            out[(size_t)row1 * OUT_F + col1] = acc11[v] + bias1;
        }
    }
}

extern "C" void kernel_launch(void* const* d_in, const int* in_sizes, int n_in,
                              void* d_out, int out_size, void* d_ws, size_t ws_size,
                              hipStream_t stream) {
    const float* X   = (const float*)d_in[0];   // logits        [B, IN_F]
    const float* Xp  = (const float*)d_in[1];   // prior_logits  [B, IN_F]
    const float* W   = (const float*)d_in[2];   // W             [E, OUT_F, IN_F]
    const float* bi  = (const float*)d_in[3];   // b             [E, OUT_F]
    const float* Wp  = (const float*)d_in[4];   // W_prior       [E, OUT_F, IN_F]
    const float* bp  = (const float*)d_in[5];   // b_prior       [E, OUT_F]
    const int*  eidx = (const int*)d_in[6];     // expert_idx    [B]
    float* out = (float*)d_out;                 // [B, 1, OUT_F]
    int*   ws  = (int*)d_ws;

    ens_zero<<<1, 64, 0, stream>>>(ws);
    ens_route<<<(B_N + 255) / 256, 256, 0, stream>>>(eidx, ws);
    ens_plan<<<1, 32, 0, stream>>>(ws);

    unsigned short* Wh  = (unsigned short*)((char*)d_ws + WS_SPLIT_OFF);
    unsigned short* Wl  = Wh + W_ELEMS;
    unsigned short* Wph = Wl + W_ELEMS;
    unsigned short* Wpl = Wph + W_ELEMS;

    const dim3 ggrid(OUT_F / N_BLOCK, MAX_TILES);
    if (ws_size >= WS_NEED) {
        const int n4 = (int)(W_ELEMS / 4);
        ens_split_w<<<(n4 + 255) / 256, 256, 0, stream>>>((const float4*)W,  (uint2*)Wh,  (uint2*)Wl,  n4);
        ens_split_w<<<(n4 + 255) / 256, 256, 0, stream>>>((const float4*)Wp, (uint2*)Wph, (uint2*)Wpl, n4);
        ens_gemm<true><<<ggrid, 256, 0, stream>>>(X, Xp, W, bi, Wp, bp, Wh, Wl, Wph, Wpl, out, ws);
    } else {
        ens_gemm<false><<<ggrid, 256, 0, stream>>>(X, Xp, W, bi, Wp, bp, Wh, Wl, Wph, Wpl, out, ws);
    }
}
